// FFNInHeadAttention_54125177864466
// MI455X (gfx1250) — matrix operand order repaired and checked by reading, not compiled, not run on hardware
//
#include <hip/hip_runtime.h>
#include <hip/hip_bf16.h>
#include <hip/hip_fp16.h>
#include <stdint.h>

typedef __attribute__((ext_vector_type(16))) __bf16 v16bf;
typedef __attribute__((ext_vector_type(8)))  float  v8f;
typedef unsigned short u16;

union Frag16 { uint4 q[2]; v16bf v; };

__device__ __forceinline__ u16 f2bf(float f) {
  uint32_t x = __float_as_uint(f);
  uint32_t r = x + 0x7fffu + ((x >> 16) & 1u);   // round-to-nearest-even
  return (u16)(r >> 16);
}

__device__ __forceinline__ v8f wmma_bf16(v16bf a, v16bf b, v8f c) {
  return __builtin_amdgcn_wmma_f32_16x16x32_bf16(false, a, false, b, (short)0, c,
                                                 false, false);
}

// A-fragment (16x32 bf16) from row-major [M][K] source.
// lane&15 = row; lo lanes: K=k0+0..7 & k0+16..23 ; hi lanes: +8.
__device__ __forceinline__ v16bf load_a_frag(const u16* __restrict__ A, int lda,
                                             int m0, int k0, int lane) {
  int l = lane & 15, hi = lane >> 4;
  const u16* p = A + (size_t)(m0 + l) * lda + k0 + hi * 8;
  Frag16 f;
  f.q[0] = *(const uint4*)(p);
  f.q[1] = *(const uint4*)(p + 16);
  return f.v;
}

// B-fragment (32x16 bf16) from B^T stored row-major [N][K].
// lane&15 = column n; lo lanes: K=k0+0..7 & k0+16..23 ; hi lanes: +8.
__device__ __forceinline__ v16bf load_b_frag(const u16* __restrict__ Bt, int ldb,
                                             int n0, int k0, int lane) {
  int l = lane & 15, hi = lane >> 4;
  const u16* p = Bt + (size_t)(n0 + l) * ldb + k0 + hi * 8;
  Frag16 f;
  f.q[0] = *(const uint4*)(p);
  f.q[1] = *(const uint4*)(p + 16);
  return f.v;
}

// Packed-f16 16-lane max reduction for two rows at once (flash attention is
// exact for any row-uniform m, so f16-rounded maxima are fine).
// Uses VOP3P v_pk_max_num_f16 directly; header __hmax2 overloads differ across
// toolchains.
__device__ __forceinline__ void rowmax_pair(float& a, float& b) {
  __half2 h = __floats2half2_rn(a, b);
  int u = __builtin_bit_cast(int, h);
#pragma unroll
  for (int m = 1; m < 16; m <<= 1) {
    int o = __shfl_xor(u, m, 32);
    asm("v_pk_max_num_f16 %0, %1, %2" : "=v"(u) : "v"(u), "v"(o));
  }
  float2 f = __half22float2(__builtin_bit_cast(__half2, u));
  a = f.x; b = f.y;
}

// ---------------------------------------------------------------------------
// GEMM: C[M,N] = A[M,K] (bf16, row-major) x B (bf16, given as B^T [N][K]) + bias
// MODE 0: +bias, exact GELU, bf16 out.  MODE 1: +bias, bf16 out.  MODE 2: +bias, f32 out.
// Block = 256 threads = 8 waves (2 along M x 4 along N); block tile 128x256,
// wave tile 64x64 = 16 WMMA accumulators -> 16 b128 loads per 16 WMMAs.
// ---------------------------------------------------------------------------
template <int MODE>
__global__ __launch_bounds__(256) void gemm_bf16_kernel(
    const u16* __restrict__ A, const u16* __restrict__ Bt,
    const float* __restrict__ bias, u16* __restrict__ outb,
    float* __restrict__ outf, int M, int N, int K) {
  const int lane = threadIdx.x & 31;
  const int wid  = threadIdx.x >> 5;
  const int wm = wid >> 2, wn = wid & 3;          // 2 x 4 wave grid
  const int bm = blockIdx.x * 128 + wm * 64;
  const int bn = blockIdx.y * 256 + wn * 64;

  v8f acc[4][4] = {};

  for (int k0 = 0; k0 < K; k0 += 32) {
    if ((k0 & 255) == 0) {
      // Speculative prefetch one 256-element stride ahead (global_prefetch_b8)
      __builtin_prefetch(A  + (size_t)(bm + lane) * K + k0 + 256, 0, 0);
      __builtin_prefetch(Bt + (size_t)(bn + lane) * K + k0 + 256, 0, 0);
    }
    v16bf a[4];
#pragma unroll
    for (int i = 0; i < 4; ++i) a[i] = load_a_frag(A, K, bm + i * 16, k0, lane);
#pragma unroll
    for (int j = 0; j < 4; ++j) {
      v16bf b = load_b_frag(Bt, K, bn + j * 16, k0, lane);
#pragma unroll
      for (int i = 0; i < 4; ++i) acc[i][j] = wmma_bf16(a[i], b, acc[i][j]);
    }
  }

  const int l = lane & 15, hi = lane >> 4;
#pragma unroll
  for (int i = 0; i < 4; ++i)
#pragma unroll
    for (int j = 0; j < 4; ++j) {
      int col = bn + j * 16 + l;
      float bv = bias[col];
#pragma unroll
      for (int r = 0; r < 8; ++r) {
        int row = bm + i * 16 + r + 8 * hi;
        float v = acc[i][j][r] + bv;
        if (MODE == 0) v = 0.5f * v * (1.0f + erff(v * 0.70710678118654752f));
        if (MODE == 2) outf[(size_t)row * N + col] = v;
        else           outb[(size_t)row * N + col] = f2bf(v);
      }
    }
}

// ---------------------------------------------------------------------------
// Flash attention, one (batch, head) per blockIdx.y, 64 query rows per block,
// 16 query rows per wave, 64 keys per chunk (P = two 16x32 WMMA A-operands).
// Denominator: per-lane partials (linearity) -> reduced once in the epilogue.
// Row max: packed-f16 pairwise shuffles -> 16 ds_bpermute per 64 keys.
// ---------------------------------------------------------------------------
__global__ __launch_bounds__(128) void attn_kernel(
    const u16* __restrict__ qb, const u16* __restrict__ kb,
    const u16* __restrict__ vt, u16* __restrict__ ob) {
  const int lane = threadIdx.x & 31;
  const int wid  = threadIdx.x >> 5;
  const int bh = blockIdx.y;
  const int b = bh >> 4, h = bh & 15;
  const int q0 = blockIdx.x * 64 + wid * 16;  // query row within batch

  const u16* qp = qb + (size_t)b * 2048 * 1024 + h * 64;  // [2048][1024] + col off
  const u16* kp = kb + (size_t)b * 2048 * 1024 + h * 64;
  const u16* vp = vt + (size_t)bh * 64 * 2048;            // V^T per head: [64][2048]

  __shared__ __align__(16) u16 plds[4][16 * 64];          // 16x64 P tile per wave
  u16* myp = plds[wid];

  const v16bf qf[2] = { load_a_frag(qp, 1024, q0, 0, lane),
                        load_a_frag(qp, 1024, q0, 32, lane) };

  v8f o[4] = {};
  float mrow[8], lrow[8];
#pragma unroll
  for (int r = 0; r < 8; ++r) { mrow[r] = -3.0e30f; lrow[r] = 0.0f; }

  const int l = lane & 15, hi = lane >> 4;
  const float scale = 0.125f;  // 1/sqrt(64)

  for (int kc = 0; kc < 2048; kc += 64) {
    // S = Q K^T for 64 keys -> four 16x16 tiles
    v8f s[4] = {};
#pragma unroll
    for (int ks = 0; ks < 2; ++ks)
#pragma unroll
      for (int t = 0; t < 4; ++t) {
        v16bf bk = load_b_frag(kp, 1024, kc + t * 16, ks * 32, lane);
        s[t] = wmma_bf16(qf[ks], bk, s[t]);
      }

    // Row maxima (row r+8*hi lives in vgpr r across the 16-lane group).
    float mx[8];
#pragma unroll
    for (int r = 0; r < 8; ++r) {
      float m01 = fmaxf(s[0][r], s[1][r]);
      float m23 = fmaxf(s[2][r], s[3][r]);
      mx[r] = fmaxf(m01, m23) * scale;
    }
#pragma unroll
    for (int r = 0; r < 8; r += 2) rowmax_pair(mx[r], mx[r + 1]);

    float alpha[8];
#pragma unroll
    for (int r = 0; r < 8; ++r) {
      float mn = fmaxf(mrow[r], mx[r]);
      alpha[r] = __expf(mrow[r] - mn);
      mrow[r] = mn;
      int row = r + 8 * hi;
      float ps = 0.0f;
#pragma unroll
      for (int t = 0; t < 4; ++t) {
        float pv = __expf(s[t][r] * scale - mn);
        ps += pv;
        myp[row * 64 + t * 16 + l] = f2bf(pv);
      }
      lrow[r] = lrow[r] * alpha[r] + ps;   // per-lane partial denominator
    }
#pragma unroll
    for (int t = 0; t < 4; ++t)
#pragma unroll
      for (int r = 0; r < 8; ++r) o[t][r] *= alpha[r];

    // C-layout -> A-layout via per-wave LDS tile. Per-wave DS ops complete in
    // order (ISA ch.11); drain before the cross-lane read for safety.
    asm volatile("s_wait_dscnt 0x0" ::: "memory");
    const u16* pl = myp + l * 64 + hi * 8;
    Frag16 f0, f1;
    f0.q[0] = *(const uint4*)(pl);       f0.q[1] = *(const uint4*)(pl + 16);
    f1.q[0] = *(const uint4*)(pl + 32);  f1.q[1] = *(const uint4*)(pl + 48);

#pragma unroll
    for (int t = 0; t < 4; ++t) {
      o[t] = wmma_bf16(f0.v, load_b_frag(vp, 2048, t * 16, kc,      lane), o[t]);
      o[t] = wmma_bf16(f1.v, load_b_frag(vp, 2048, t * 16, kc + 32, lane), o[t]);
    }
    // Order this iteration's LDS reads before next iteration's stores
    // (hardware keeps per-wave DS in order; compiler-only fence).
    asm volatile("" ::: "memory");
  }

  // Final 16-lane reduction of the softmax denominators.
#pragma unroll
  for (int r = 0; r < 8; ++r) {
    float s = lrow[r];
#pragma unroll
    for (int m = 1; m < 16; m <<= 1) s += __shfl_xor(s, m, 32);
    lrow[r] = s;
  }

#pragma unroll
  for (int t = 0; t < 4; ++t) {
    int col = h * 64 + t * 16 + l;
#pragma unroll
    for (int r = 0; r < 8; ++r) {
      size_t row = (size_t)b * 2048 + q0 + r + 8 * hi;
      ob[row * 1024 + col] = f2bf(o[t][r] / lrow[r]);
    }
  }
}

// ---------------------------------------------------------------------------
// Data-prep kernels (memory-bound, tiny vs. the GEMMs)
// ---------------------------------------------------------------------------
__global__ void cvt_f32_bf16(const float* __restrict__ in, u16* __restrict__ out,
                             size_t n) {
  size_t stride = (size_t)gridDim.x * blockDim.x;
  for (size_t i = (size_t)blockIdx.x * blockDim.x + threadIdx.x; i < n; i += stride)
    out[i] = f2bf(in[i]);
}

// in: [K][N] f32 row-major  ->  out: [N][K] bf16 row-major (i.e. B^T)
__global__ void transpose_cvt(const float* __restrict__ in, u16* __restrict__ out,
                              int K, int N) {
  size_t n = (size_t)K * N;
  size_t stride = (size_t)gridDim.x * blockDim.x;
  for (size_t i = (size_t)blockIdx.x * blockDim.x + threadIdx.x; i < n; i += stride) {
    int k = (int)(i / (size_t)N), c = (int)(i % (size_t)N);
    out[(size_t)c * K + k] = f2bf(in[i]);
  }
}

// v [B,N,C] bf16 -> vT per head: vT[(bh*64+d)*2048 + n] = v[(b*2048+n)*1024 + h*64+d]
__global__ void build_vT(const u16* __restrict__ v, u16* __restrict__ vt) {
  size_t n = (size_t)4096 * 2048;  // (B*H*64) rows x 2048
  size_t stride = (size_t)gridDim.x * blockDim.x;
  for (size_t i = (size_t)blockIdx.x * blockDim.x + threadIdx.x; i < n; i += stride) {
    int col = (int)(i & 2047);
    size_t row = i >> 11;
    int d = (int)(row & 63);
    int bh = (int)(row >> 6);
    int h = bh & 15, b = bh >> 4;
    vt[i] = v[((size_t)b * 2048 + col) * 1024 + h * 64 + d];
  }
}

// ---------------------------------------------------------------------------
extern "C" void kernel_launch(void* const* d_in, const int* in_sizes, int n_in,
                              void* d_out, int out_size, void* d_ws, size_t ws_size,
                              hipStream_t stream) {
  (void)in_sizes; (void)n_in; (void)out_size; (void)ws_size;
  const float* x  = (const float*)d_in[0];
  const float* w1[3] = { (const float*)d_in[1], (const float*)d_in[5], (const float*)d_in[9]  };
  const float* b1[3] = { (const float*)d_in[2], (const float*)d_in[6], (const float*)d_in[10] };
  const float* w2[3] = { (const float*)d_in[3], (const float*)d_in[7], (const float*)d_in[11] };
  const float* b2[3] = { (const float*)d_in[4], (const float*)d_in[8], (const float*)d_in[12] };
  const float* pw = (const float*)d_in[13];
  const float* pb = (const float*)d_in[14];
  float* out = (float*)d_out;

  constexpr size_t M = 8192, C = 1024, HID = 4096;

  u16* p = (u16*)d_ws;
  u16* xb = p;                 p += M * C;
  u16* w1t[3]; for (int i = 0; i < 3; ++i) { w1t[i] = p; p += HID * C; }
  u16* w2t[3]; for (int i = 0; i < 3; ++i) { w2t[i] = p; p += C * HID; }
  u16* projt = p;              p += C * C;
  u16* hbuf = p;               p += M * HID;       // reused for q/k/v fc1 output
  u16* qkv[3]; for (int i = 0; i < 3; ++i) { qkv[i] = p; p += M * C; }
  u16* vtb = p;                p += (size_t)4096 * 2048;
  u16* attnb = p;              p += M * C;

  // bf16 conversions / weight transposes
  cvt_f32_bf16<<<2048, 256, 0, stream>>>(x, xb, M * C);
  for (int i = 0; i < 3; ++i) {
    transpose_cvt<<<8192, 256, 0, stream>>>(w1[i], w1t[i], 1024, 4096);
    transpose_cvt<<<8192, 256, 0, stream>>>(w2[i], w2t[i], 4096, 1024);
  }
  transpose_cvt<<<2048, 256, 0, stream>>>(pw, projt, 1024, 1024);

  // q/k/v MLPs: fc1(+GELU) then fc2(+bias)
  for (int i = 0; i < 3; ++i) {
    dim3 g1(8192 / 128, 4096 / 256);
    gemm_bf16_kernel<0><<<g1, 256, 0, stream>>>(xb, w1t[i], b1[i], hbuf, nullptr,
                                                8192, 4096, 1024);
    dim3 g2(8192 / 128, 1024 / 256);
    gemm_bf16_kernel<1><<<g2, 256, 0, stream>>>(hbuf, w2t[i], b2[i], qkv[i], nullptr,
                                                8192, 1024, 4096);
  }

  // attention
  build_vT<<<8192, 256, 0, stream>>>(qkv[2], vtb);
  attn_kernel<<<dim3(32, 64), 128, 0, stream>>>(qkv[0], qkv[1], vtb, attnb);

  // output projection (f32 out)
  dim3 g3(8192 / 128, 1024 / 256);
  gemm_bf16_kernel<2><<<g3, 256, 0, stream>>>(attnb, projt, pb, nullptr, out,
                                              8192, 1024, 1024);
}
